// CausalSelfAttention_549755814137
// MI455X (gfx1250) — matrix operand order repaired and checked by reading, not compiled, not run on hardware
//
#include <hip/hip_runtime.h>

// ---------------- problem constants ----------------
#define BB 2
#define TT 4096
#define CC 768
#define HH 12
#define HDIM 64
#define MROWS (BB * TT)   // 8192 token rows
#define NQKV (3 * CC)     // 2304

// ---------------- types ----------------
typedef __bf16 bf16_t;
typedef bf16_t bf16x16 __attribute__((ext_vector_type(16)));
typedef float f32x8 __attribute__((ext_vector_type(8)));
typedef unsigned int u32x4 __attribute__((ext_vector_type(4)));

union Frag {
  bf16x16 b;
  u32x4 q[2];
  unsigned short us[16];
};

__device__ __forceinline__ u32x4 ld128(const unsigned short* p) {
  return *(const u32x4*)p;
}

// float -> bf16 bits, round-to-nearest-even, NaN-safe
__device__ __forceinline__ unsigned short f2bf(float f) {
  union { float f; unsigned u; } c;
  c.f = f;
  unsigned u = c.u;
  if ((u & 0x7F800000u) == 0x7F800000u && (u & 0x007FFFFFu))
    return (unsigned short)((u >> 16) | 0x40);  // quiet NaN
  u += 0x7FFFu + ((u >> 16) & 1u);
  return (unsigned short)(u >> 16);
}

__device__ __forceinline__ f32x8 wmma_bf16(const Frag& a, const Frag& b, f32x8 c) {
  return __builtin_amdgcn_wmma_f32_16x16x32_bf16(false, a.b, false, b.b,
                                                 (short)0, c, false, false);
}

// raw LDS byte offset of a __shared__ object (for async-to-LDS dest operand)
__device__ __forceinline__ unsigned lds_offset(const void* p) {
  return (unsigned)(unsigned long long)(const __attribute__((address_space(3))) char*)p;
}

// one per-lane 16B async copy global -> LDS (ASYNCcnt-tracked)
__device__ __forceinline__ void async_cp16(unsigned lds_byte, unsigned long long gaddr) {
  asm volatile("global_load_async_to_lds_b128 %0, %1, off"
               :: "v"(lds_byte), "v"(gaddr)
               : "memory");
}

// ---------------- conversion kernels ----------------
__global__ void __launch_bounds__(256) k_cvt_bf16(const float* __restrict__ in,
                                                  unsigned short* __restrict__ out,
                                                  int n) {
  int i = blockIdx.x * 256 + threadIdx.x;
  if (i < n) out[i] = f2bf(in[i]);
}

// in: [K][N] f32 row-major  ->  out: [N][K] bf16 (transposed)
__global__ void __launch_bounds__(256) k_transpose_bf16(const float* __restrict__ in,
                                                        unsigned short* __restrict__ out,
                                                        int K, int N) {
  int i = blockIdx.x * 256 + threadIdx.x;
  if (i < K * N) {
    int k = i / N, n = i % N;
    out[n * K + k] = f2bf(in[i]);
  }
}

// ---------------- QKV GEMM: [8192,768] x [768,2304] -> Q,K,V^T (bf16) --------
// 32x64 tile per wave, double-buffered K loop (k+32 in flight during compute).
__global__ void __launch_bounds__(256) k_gemm_qkv(
    const unsigned short* __restrict__ A, const unsigned short* __restrict__ WT,
    const float* __restrict__ bias, unsigned short* __restrict__ Qo,
    unsigned short* __restrict__ Ko, unsigned short* __restrict__ Vt) {
  const int NTN = NQKV / 64;  // 36 N-tiles of 64
  int wave = blockIdx.x * 8 + (threadIdx.x >> 5);
  int mt = wave / NTN, nt = wave % NTN;
  int m0 = mt * 32, n0 = nt * 64;
  int lane = threadIdx.x & 31, lo = lane & 15, hi = lane >> 4;

  f32x8 zero = {0.f, 0.f, 0.f, 0.f, 0.f, 0.f, 0.f, 0.f};
  f32x8 acc[2][4] = {{zero, zero, zero, zero}, {zero, zero, zero, zero}};

  const unsigned short* Arow0 = A + (m0 + lo) * CC;
  const unsigned short* Arow1 = A + (m0 + 16 + lo) * CC;

  Frag aA[2], aB[2], bA[4], bB[4];

  auto loadA = [&](Frag* a, int k0) {
    a[0].q[0] = ld128(Arow0 + k0 + 8 * hi);
    a[0].q[1] = ld128(Arow0 + k0 + 16 + 8 * hi);
    a[1].q[0] = ld128(Arow1 + k0 + 8 * hi);
    a[1].q[1] = ld128(Arow1 + k0 + 16 + 8 * hi);
  };
  auto loadB = [&](Frag* b, int k0) {
#pragma unroll
    for (int s = 0; s < 4; ++s) {
      const unsigned short* Brow = WT + (n0 + s * 16 + lo) * CC + k0 + 8 * hi;
      b[s].q[0] = ld128(Brow);
      b[s].q[1] = ld128(Brow + 16);
    }
  };
  auto mma8 = [&](Frag* a, Frag* b) {
#pragma unroll
    for (int s = 0; s < 4; ++s) {
      acc[0][s] = wmma_bf16(a[0], b[s], acc[0][s]);
      acc[1][s] = wmma_bf16(a[1], b[s], acc[1][s]);
    }
  };

  loadA(aA, 0);
  loadB(bA, 0);
#pragma unroll 2
  for (int k0 = 0; k0 < CC; k0 += 64) {
    loadA(aB, k0 + 32);  // in flight while computing k0
    loadB(bB, k0 + 32);
    mma8(aA, bA);
    int kn = (k0 + 64 < CC) ? k0 + 64 : 0;  // clamped redundant prefetch on last iter
    loadA(aA, kn);
    loadB(bA, kn);
    mma8(aB, bB);
  }

#pragma unroll
  for (int u = 0; u < 2; ++u) {
#pragma unroll
    for (int s = 0; s < 4; ++s) {
      int col = n0 + s * 16 + lo;  // 0..2303
      float bv = bias[col];
      int seg = col / CC;  // 0=Q 1=K 2=V (uniform per s-tile)
      int c = col - seg * CC;
      int h = c >> 6, hd = c & 63;
#pragma unroll
      for (int r = 0; r < 8; ++r) {
        int row = m0 + u * 16 + r + 8 * hi;  // token row
        int b = row >> 12, t = row & (TT - 1);
        unsigned short val = f2bf(acc[u][s][r] + bv);
        int bh = b * HH + h;
        if (seg == 0)
          Qo[(bh * TT + t) * HDIM + hd] = val;
        else if (seg == 1)
          Ko[(bh * TT + t) * HDIM + hd] = val;
        else
          Vt[(bh * HDIM + hd) * TT + t] = val;
      }
    }
  }
}

// ---------------- flash attention: one wave per (b,h,16-query tile) ----------
// K/V tiles are DMAed into per-wave LDS with async-to-LDS (double-buffered);
// fragments are built from LDS; P is re-laid-out D->A through LDS.
#define AW 2  // waves per block
// per-wave LDS map (ushort indices): K0 @0, K1 @2048, V0 @4096, V1 @6144, P @8192
#define WSLOT 8704

__global__ void __launch_bounds__(AW * 32) k_attn(const unsigned short* __restrict__ Qm,
                                                  const unsigned short* __restrict__ Km,
                                                  const unsigned short* __restrict__ Vt,
                                                  unsigned short* __restrict__ Y) {
  __shared__ __align__(16) unsigned short lds[AW][WSLOT];
  int wv = threadIdx.x >> 5;
  int wave = blockIdx.x * AW + wv;
  int qt = wave & 255;  // 256 query tiles
  int bh = wave >> 8;   // 0..23
  int q0 = qt * 16;
  int lane = threadIdx.x & 31, lo = lane & 15, hi = lane >> 4;

  const unsigned short* Qb = Qm + bh * (TT * HDIM);
  const unsigned short* Kb = Km + bh * (TT * HDIM);
  const unsigned short* Vb = Vt + bh * (HDIM * TT);

  unsigned short* Ktile[2] = {&lds[wv][0], &lds[wv][2048]};
  unsigned short* Vtile[2] = {&lds[wv][4096], &lds[wv][6144]};
  unsigned short* Pp = &lds[wv][8192];
  unsigned waveBase = lds_offset(&lds[wv][0]);  // LDS byte offset
  unsigned kBase[2] = {waveBase, waveBase + 4096u};
  unsigned vBase[2] = {waveBase + 8192u, waveBase + 12288u};

  // K tile rows are contiguous in memory: pure 4KB linear async copy
  auto copyK = [&](unsigned dstB, int kb) {
    unsigned long long g =
        (unsigned long long)(const void*)(Kb + kb * HDIM) + (unsigned)lane * 16u;
#pragma unroll
    for (int i = 0; i < 8; ++i)
      async_cp16(dstB + (unsigned)(lane * 16 + i * 512), g + (unsigned)(i * 512));
  };
  // V^T tile: 64 rows (hd), 32 keys each (64B), row stride T*2 bytes
  auto copyV = [&](unsigned dstB, int kb) {
    int row = lane >> 2;        // base row 0..7
    int colB = (lane & 3) * 16; // byte offset within 64B row
    unsigned long long g = (unsigned long long)(const void*)Vb +
                           (unsigned long long)kb * 2u + (unsigned)colB;
#pragma unroll
    for (int i = 0; i < 8; ++i) {
      int rr = i * 8 + row;
      async_cp16(dstB + (unsigned)(rr * 64 + colB),
                 g + (unsigned long long)rr * (TT * 2));
    }
  };

  Frag qf0, qf1;  // Q fragments for k-chunks 0..31 and 32..63
  {
    const unsigned short* qrow = Qb + (q0 + lo) * HDIM;
    qf0.q[0] = ld128(qrow + 16 * hi);
    qf0.q[1] = ld128(qrow + 16 * hi + 8);
    qf1.q[0] = ld128(qrow + 32 + 16 * hi);
    qf1.q[1] = ld128(qrow + 32 + 16 * hi + 8);
  }

  f32x8 zero = {0.f, 0.f, 0.f, 0.f, 0.f, 0.f, 0.f, 0.f};
  f32x8 o[4] = {zero, zero, zero, zero};
  float m_i[8], l_i[8];
#pragma unroll
  for (int r = 0; r < 8; ++r) {
    m_i[r] = -__builtin_inff();
    l_i[r] = 0.f;
  }

  const float scale = 0.125f;  // 1/sqrt(64)
  int nIter = (q0 + 16 + 31) >> 5;

  copyK(kBase[0], 0);
  copyV(vBase[0], 0);

  int cur = 0;
  for (int it = 0; it < nIter; ++it) {
    int kb = it * 32;
    int knb = kb + 32;
    if (knb > TT - 32) knb = TT - 32;  // clamped; over-read data is masked anyway
    copyK(kBase[cur ^ 1], knb);        // next tile in flight during compute
    copyV(vBase[cur ^ 1], knb);

    // first 16 async ops (current tile) retired; 16 (next tile) may remain
    asm volatile("s_wait_asynccnt 0x10" ::: "memory");

    const unsigned short* Kt = Ktile[cur];
    const unsigned short* Vtl = Vtile[cur];

    // ---- S = Q K^T for 32 keys (two 16-key N-tiles), K frags from LDS ----
    Frag k0f, k1f, k2f, k3f;
    {
      const unsigned short* kp0 = Kt + lo * 64 + 16 * hi;
      const unsigned short* kp1 = Kt + (16 + lo) * 64 + 16 * hi;
      k0f.q[0] = ld128(kp0);       k0f.q[1] = ld128(kp0 + 8);
      k1f.q[0] = ld128(kp0 + 32);  k1f.q[1] = ld128(kp0 + 40);
      k2f.q[0] = ld128(kp1);       k2f.q[1] = ld128(kp1 + 8);
      k3f.q[0] = ld128(kp1 + 32);  k3f.q[1] = ld128(kp1 + 40);
    }
    f32x8 s0 = zero, s1 = zero;
    s0 = wmma_bf16(qf0, k0f, s0);
    s0 = wmma_bf16(qf1, k1f, s0);
    s1 = wmma_bf16(qf0, k2f, s1);
    s1 = wmma_bf16(qf1, k3f, s1);

    // ---- scale + causal mask + online softmax ----
    int c0 = kb + lo, c1 = kb + 16 + lo;
#pragma unroll
    for (int r = 0; r < 8; ++r) {
      int row = q0 + r + 8 * hi;
      float v0 = (c0 <= row) ? s0[r] * scale : -__builtin_inff();
      float v1 = (c1 <= row) ? s1[r] * scale : -__builtin_inff();
      float mx = fmaxf(v0, v1);
#pragma unroll
      for (int off = 8; off >= 1; off >>= 1)
        mx = fmaxf(mx, __shfl_xor(mx, off, 32));
      float mn = fmaxf(m_i[r], mx);
      float alpha = __expf(m_i[r] - mn);
      float p0 = __expf(v0 - mn);
      float p1 = __expf(v1 - mn);
      float rs = p0 + p1;
#pragma unroll
      for (int off = 8; off >= 1; off >>= 1)
        rs += __shfl_xor(rs, off, 32);
      l_i[r] = l_i[r] * alpha + rs;
      m_i[r] = mn;
      o[0][r] *= alpha;
      o[1][r] *= alpha;
      o[2][r] *= alpha;
      o[3][r] *= alpha;
      int M = r + 8 * hi;  // D-layout row -> LDS row
      Pp[M * 32 + lo] = f2bf(p0);
      Pp[M * 32 + 16 + lo] = f2bf(p1);
    }

    // LDS is in-order per wave; fence hw + compiler before re-reading P
    asm volatile("s_wait_dscnt 0" ::: "memory");

    Frag pf;  // P as A-fragment (M=lane&15, contiguous k = 16*hi + s, same as V frags)
    {
      const unsigned short* prow = Pp + lo * 32;
      pf.q[0] = *(const u32x4*)(prow + 16 * hi);
      pf.q[1] = *(const u32x4*)(prow + 16 * hi + 8);
    }

    // ---- O += P V, V frags from LDS ----
#pragma unroll
    for (int s = 0; s < 4; ++s) {
      const unsigned short* vrow = Vtl + (s * 16 + lo) * 32 + 16 * hi;
      Frag bv;
      bv.q[0] = ld128(vrow);
      bv.q[1] = ld128(vrow + 8);
      o[s] = wmma_bf16(pf, bv, o[s]);
    }

    cur ^= 1;
  }

  // ---- normalize and store y (bf16, token-major [B*T, C]) ----
  int b = bh / HH, h = bh % HH;
#pragma unroll
  for (int r = 0; r < 8; ++r) {
    float inv = 1.0f / l_i[r];
    int row = q0 + r + 8 * hi;
    unsigned base = (unsigned)(b * TT + row) * CC + h * HDIM;
#pragma unroll
    for (int s = 0; s < 4; ++s)
      Y[base + s * 16 + lo] = f2bf(o[s][r] * inv);
  }
}

// ---------------- proj GEMM: [8192,768] x [768,768] + bias -> f32 out --------
__global__ void __launch_bounds__(256) k_gemm_proj(
    const unsigned short* __restrict__ A, const unsigned short* __restrict__ WT,
    const float* __restrict__ bias, float* __restrict__ Out) {
  const int NTN = CC / 64;  // 12
  int wave = blockIdx.x * 8 + (threadIdx.x >> 5);
  int mt = wave / NTN, nt = wave % NTN;
  int m0 = mt * 32, n0 = nt * 64;
  int lane = threadIdx.x & 31, lo = lane & 15, hi = lane >> 4;

  f32x8 zero = {0.f, 0.f, 0.f, 0.f, 0.f, 0.f, 0.f, 0.f};
  f32x8 acc[2][4] = {{zero, zero, zero, zero}, {zero, zero, zero, zero}};

  const unsigned short* Arow0 = A + (m0 + lo) * CC;
  const unsigned short* Arow1 = A + (m0 + 16 + lo) * CC;

  Frag aA[2], aB[2], bA[4], bB[4];

  auto loadA = [&](Frag* a, int k0) {
    a[0].q[0] = ld128(Arow0 + k0 + 8 * hi);
    a[0].q[1] = ld128(Arow0 + k0 + 16 + 8 * hi);
    a[1].q[0] = ld128(Arow1 + k0 + 8 * hi);
    a[1].q[1] = ld128(Arow1 + k0 + 16 + 8 * hi);
  };
  auto loadB = [&](Frag* b, int k0) {
#pragma unroll
    for (int s = 0; s < 4; ++s) {
      const unsigned short* Brow = WT + (n0 + s * 16 + lo) * CC + k0 + 8 * hi;
      b[s].q[0] = ld128(Brow);
      b[s].q[1] = ld128(Brow + 16);
    }
  };
  auto mma8 = [&](Frag* a, Frag* b) {
#pragma unroll
    for (int s = 0; s < 4; ++s) {
      acc[0][s] = wmma_bf16(a[0], b[s], acc[0][s]);
      acc[1][s] = wmma_bf16(a[1], b[s], acc[1][s]);
    }
  };

  loadA(aA, 0);
  loadB(bA, 0);
#pragma unroll 2
  for (int k0 = 0; k0 < CC; k0 += 64) {
    loadA(aB, k0 + 32);
    loadB(bB, k0 + 32);
    mma8(aA, bA);
    int kn = (k0 + 64 < CC) ? k0 + 64 : 0;
    loadA(aA, kn);
    loadB(bA, kn);
    mma8(aB, bB);
  }

#pragma unroll
  for (int u = 0; u < 2; ++u) {
#pragma unroll
    for (int s = 0; s < 4; ++s) {
      int col = n0 + s * 16 + lo;
      float bv = bias[col];
#pragma unroll
      for (int r = 0; r < 8; ++r) {
        int row = m0 + u * 16 + r + 8 * hi;
        Out[(unsigned)row * CC + col] = acc[u][s][r] + bv;
      }
    }
  }
}

// ---------------- host launcher ----------------
extern "C" void kernel_launch(void* const* d_in, const int* in_sizes, int n_in,
                              void* d_out, int out_size, void* d_ws, size_t ws_size,
                              hipStream_t stream) {
  const float* x = (const float*)d_in[0];       // [B,T,C]
  const float* w_qkv = (const float*)d_in[1];   // [C,3C]
  const float* b_qkv = (const float*)d_in[2];   // [3C]
  const float* w_proj = (const float*)d_in[3];  // [C,C]
  const float* b_proj = (const float*)d_in[4];  // [C]
  float* out = (float*)d_out;                   // [B,T,C] f32

  char* ws = (char*)d_ws;
  size_t off = 0;
  auto take = [&](size_t bytes) {
    void* p = ws + off;
    off = (off + bytes + 255) & ~(size_t)255;
    return p;
  };
  unsigned short* xb = (unsigned short*)take((size_t)MROWS * CC * 2);
  unsigned short* wqkvT = (unsigned short*)take((size_t)NQKV * CC * 2);
  unsigned short* wprojT = (unsigned short*)take((size_t)CC * CC * 2);
  unsigned short* Qh = (unsigned short*)take((size_t)BB * HH * TT * HDIM * 2);
  unsigned short* Kh = (unsigned short*)take((size_t)BB * HH * TT * HDIM * 2);
  unsigned short* Vt = (unsigned short*)take((size_t)BB * HH * TT * HDIM * 2);
  unsigned short* Yb = (unsigned short*)take((size_t)MROWS * CC * 2);

  {
    int n = MROWS * CC;
    k_cvt_bf16<<<(n + 255) / 256, 256, 0, stream>>>(x, xb, n);
  }
  {
    int n = CC * NQKV;
    k_transpose_bf16<<<(n + 255) / 256, 256, 0, stream>>>(w_qkv, wqkvT, CC, NQKV);
  }
  {
    int n = CC * CC;
    k_transpose_bf16<<<(n + 255) / 256, 256, 0, stream>>>(w_proj, wprojT, CC, CC);
  }
  {
    int waves = (MROWS / 32) * (NQKV / 64);  // 256*36 = 9216
    k_gemm_qkv<<<waves / 8, 256, 0, stream>>>(xb, wqkvT, b_qkv, Qh, Kh, Vt);
  }
  {
    int waves = BB * HH * (TT / 16);  // 6144
    k_attn<<<waves / AW, AW * 32, 0, stream>>>(Qh, Kh, Vt, Yb);
  }
  {
    int waves = (MROWS / 32) * (CC / 64);  // 3072
    k_gemm_proj<<<waves / 8, 256, 0, stream>>>(Yb, wprojT, b_proj, out);
  }
}
